// MultiHeadAttention_43155831390327
// MI455X (gfx1250) — matrix operand order repaired and checked by reading, not compiled, not run on hardware
//
#include <hip/hip_runtime.h>
#include <hip/hip_bf16.h>

// ---------------------------------------------------------------------------
// MI455X (gfx1250, wave32) fused MHA with memory slots.
// All matmuls via v_wmma_f32_16x16x32_f16; tile staging via
// global_load_async_to_lds_b128 (ASYNCcnt) with double buffering;
// softmax reductions via DPP (no ds_bpermute round trips).
// ---------------------------------------------------------------------------

typedef _Float16 h16;
typedef __attribute__((ext_vector_type(16))) _Float16 v16h;
typedef __attribute__((ext_vector_type(8)))  float    v8f;
typedef __attribute__((ext_vector_type(4)))  unsigned int u32x4;

union V16 {
    v16h  v;
    h16   h[16];
    u32x4 q[2];
};

static __device__ __forceinline__ v8f wmma_f16(v16h a, v16h b, v8f c) {
    return __builtin_amdgcn_wmma_f32_16x16x32_f16(
        /*neg_a=*/false, a, /*neg_b=*/false, b,
        /*c_mod=*/(short)0, c, /*reuse_a=*/false, /*reuse_b=*/false);
}

// Async 16B global->LDS copy (ASYNCcnt-tracked). VDST carries the per-lane
// LDS byte offset (low 32 bits of the flat shared address), VADDR the global VA.
static __device__ __forceinline__ void async_b128(void* lds, const void* g) {
    asm volatile("global_load_async_to_lds_b128 %0, %1, off"
                 :: "v"((unsigned)(uintptr_t)lds),
                    "v"((unsigned long long)(uintptr_t)g)
                 : "memory");
}
#define ASYNC_WAIT(imm) asm volatile("s_wait_asynccnt " #imm ::: "memory")

// ---- DPP lane reductions over 16-lane groups (wave32 WMMA N-striping) ----
#define SEL8(a,b,c,d,e,f,g,h) \
    ((a)|((b)<<3)|((c)<<6)|((d)<<9)|((e)<<12)|((f)<<15)|((g)<<18)|((h)<<21))
#define DPP8_XOR1 SEL8(1,0,3,2,5,4,7,6)
#define DPP8_XOR2 SEL8(2,3,0,1,6,7,4,5)
#define DPP8_XOR4 SEL8(4,5,6,7,0,1,2,3)

template <int SEL>
static __device__ __forceinline__ float dpp8f(float x) {
    return __builtin_bit_cast(float,
        __builtin_amdgcn_mov_dpp8(__builtin_bit_cast(int, x), SEL));
}
static __device__ __forceinline__ float dpp16_mirror(float x) {
    // DPP16 ROW_MIRROR (0x140): lane <-> 15-lane within each 16-lane row
    return __builtin_bit_cast(float,
        __builtin_amdgcn_update_dpp(0, __builtin_bit_cast(int, x),
                                    0x140, 0xf, 0xf, true));
}
static __device__ __forceinline__ float red_max16(float x) {
    x = fmaxf(x, dpp8f<DPP8_XOR1>(x));
    x = fmaxf(x, dpp8f<DPP8_XOR2>(x));
    x = fmaxf(x, dpp8f<DPP8_XOR4>(x));
    x = fmaxf(x, dpp16_mirror(x));       // combines the two 8-lane halves
    return x;
}
static __device__ __forceinline__ float red_sum16(float x) {
    x += dpp8f<DPP8_XOR1>(x);
    x += dpp8f<DPP8_XOR2>(x);
    x += dpp8f<DPP8_XOR4>(x);
    x += dpp16_mirror(x);
    return x;
}

#define D_MODEL 1024
#define NHEAD   16
#define DK      64
#define NQ      1024
#define BATCH   8
#define MSLOT   8
#define NKE     1056   // 1024 real keys + 8 memory slots + 24 zero pad = 33*32

// --------------------------- elementwise kernels ---------------------------

__global__ void cast_f32_f16(const float* __restrict__ s, h16* __restrict__ d, int n) {
    int i = blockIdx.x * 256 + threadIdx.x;
    if (i < n) d[i] = (h16)s[i];
}

// (NQ,B,D) fp32 -> (B*NQ, D) fp16  (folds the seq/batch transpose)
__global__ void cast_act(const float* __restrict__ x, h16* __restrict__ y) {
    int i = blockIdx.x * 256 + threadIdx.x;   // 8192*1024 threads
    int dd = i & 1023;
    int r  = i >> 10;                          // out row = b*NQ + n
    int b  = r >> 10;
    int n  = r & 1023;
    y[(size_t)i] = (h16)x[(size_t)n * (BATCH * D_MODEL) + (size_t)b * D_MODEL + dd];
}

// rows 1024..1055 of padded K/V: 8 scaled memory slots + 24 zero rows
__global__ void fill_mem(const float* __restrict__ mk, const float* __restrict__ mv,
                         h16* __restrict__ kx, h16* __restrict__ vx) {
    int i  = blockIdx.x * 256 + threadIdx.x;   // 8*32*1024 threads
    int dd = i & 1023;
    int row = i >> 10;
    int rl = row & 31;
    int b  = row >> 5;
    h16 kv = (h16)0.f, vv = (h16)0.f;
    if (rl < MSLOT) {
        kv = (h16)(32.0f * mk[rl * D_MODEL + dd]);         // sqrt(DK*H)=32
        vv = (h16)(2.8284271247f * mv[rl * D_MODEL + dd]); // sqrt(M)
    }
    size_t o = ((size_t)b * NKE + 1024 + rl) * D_MODEL + dd;
    kx[o] = kv;
    vx[o] = vv;
}

// ------------------------------ WMMA GEMM ----------------------------------
// C[M,1024] = A[M,1024] @ W^T + bias    (A,W fp16 row-major, W is [N,K])
// Double-buffered async LDS staging: tile k+1 streams while tile k computes.
// MODE 0: out fp16 row-major (Q)
// MODE 1: out fp16 with per-batch row pad 1024->1056 (K/V into padded bufs)
// MODE 2: out fp32, rows remapped (b,n)->(n,b) for final (seq,batch,d) output
template <int MODE>
__global__ void gemm_wmma(const h16* __restrict__ A, const h16* __restrict__ W,
                          const float* __restrict__ bias, void* __restrict__ Out) {
    __shared__ __align__(16) h16 As[2][128 * 32];
    __shared__ __align__(16) h16 Bs[2][128 * 32];
    const int t    = threadIdx.x;
    const int lane = t & 31, wave = t >> 5;
    const int wm = wave >> 2, wn = wave & 3;      // 2x4 wave grid -> 128x128 tile
    const int mbase = blockIdx.x * 128, nbase = blockIdx.y * 128;
    const int lm = lane & 15;

    const int row = t >> 1, kh = (t & 1) * 16;    // staging role: 32B of A + 32B of B
    auto issue_tile = [&](int buf, int kb) {      // 4 async b128 ops / thread
        const h16* sa = A + (size_t)(mbase + row) * D_MODEL + kb + kh;
        async_b128(&As[buf][row * 32 + kh],     sa);
        async_b128(&As[buf][row * 32 + kh + 8], sa + 8);
        const h16* sb = W + (size_t)(nbase + row) * D_MODEL + kb + kh;
        async_b128(&Bs[buf][row * 32 + kh],     sb);
        async_b128(&Bs[buf][row * 32 + kh + 8], sb + 8);
    };

    v8f acc[4][2] = {};
    issue_tile(0, 0);   // prologue prefetch

    for (int it = 0; it < 32; ++it) {
        __syncthreads();                       // readers done with buf[(it+1)&1]
        if (it + 1 < 32) {
            issue_tile((it + 1) & 1, (it + 1) * 32);
            ASYNC_WAIT(0x4);                   // drain current tile, keep next in flight
        } else {
            ASYNC_WAIT(0x0);
        }
        __syncthreads();                       // publish buf[it&1] to all waves

        const h16* as = As[it & 1];
        const h16* bs = Bs[it & 1];
        V16 bv[2];
        {   // B operand: lane=col n, halves = K {kk0..kk0+7, kk0+16..kk0+23}
            int kk0 = (lane < 16) ? 0 : 8;
            #pragma unroll
            for (int cb = 0; cb < 2; ++cb) {
                const u32x4* p = (const u32x4*)(bs + (wn * 32 + cb * 16 + lm) * 32 + kk0);
                bv[cb].q[0] = p[0]; bv[cb].q[1] = p[2];
            }
        }
        int k0 = (lane < 16) ? 0 : 8;  // A layout: halves {k0..k0+7, k0+16..k0+23}
        #pragma unroll
        for (int rb = 0; rb < 4; ++rb) {
            V16 av;
            int arow = wm * 64 + rb * 16 + lm;
            av.q[0] = *(const u32x4*)(as + arow * 32 + k0);
            av.q[1] = *(const u32x4*)(as + arow * 32 + k0 + 16);
            #pragma unroll
            for (int cb = 0; cb < 2; ++cb)
                acc[rb][cb] = wmma_f16(av.v, bv[cb].v, acc[rb][cb]);
        }
    }

    const int lh = lane >> 4;
    #pragma unroll
    for (int rb = 0; rb < 4; ++rb)
        #pragma unroll
        for (int cb = 0; cb < 2; ++cb)
            #pragma unroll
            for (int r = 0; r < 8; ++r) {
                int grow = mbase + wm * 64 + rb * 16 + r + lh * 8;
                int gcol = nbase + wn * 32 + cb * 16 + lm;
                float v = acc[rb][cb][r] + bias[gcol];
                if (MODE == 0) {
                    ((h16*)Out)[(size_t)grow * D_MODEL + gcol] = (h16)v;
                } else if (MODE == 1) {
                    int orow = (grow >> 10) * NKE + (grow & 1023);
                    ((h16*)Out)[(size_t)orow * D_MODEL + gcol] = (h16)v;
                } else {
                    int orow = ((grow & 1023) << 3) + (grow >> 10); // (b,n)->(n,b)
                    ((float*)Out)[(size_t)orow * D_MODEL + gcol] = v;
                }
            }
}

// --------------------------- flash attention -------------------------------
// grid (NQ/128, H, B), block 256 (8 waves). Wave owns 16 q-rows.
// K tile + mask tile double-buffered via async copies (the 512MB mask stream
// is the dominant HBM traffic -> prefetch one tile ahead, read from LDS).
__global__ void attn_flash(const h16* __restrict__ Q, const h16* __restrict__ K,
                           const h16* __restrict__ V, const float* __restrict__ mask,
                           h16* __restrict__ O) {
    __shared__ __align__(16) h16   Kbuf[2][32 * 64];   // [key][dk] row-major
    __shared__ __align__(16) float Mbuf[2][128 * 32];  // [qlocal][key] fp32 mask tile
    __shared__ __align__(16) h16   Vt[64 * 40];        // [dv][key] transposed, 16B rows
    __shared__ __align__(16) h16   Pex[8 * 16 * 40];   // per-wave P exchange, 16B rows

    const int t = threadIdx.x, lane = t & 31, wave = t >> 5;
    const int qblk = blockIdx.x, h = blockIdx.y, b = blockIdx.z;
    const int qbase = qblk * 128 + wave * 16;
    const int lm = lane & 15, lh = lane >> 4;

    // async staging roles
    const int skey = t >> 3, soff = (t & 7) * 8;        // K: 16B / thread
    const int mrow = t >> 1, mseg = (t & 1) * 16;       // mask: 64B / thread
    auto issue_k = [&](int buf, int kbase) {            // 1 op
        const h16* src = K + (size_t)(b * NKE + kbase + skey) * D_MODEL + h * DK + soff;
        async_b128(&Kbuf[buf][skey * 64 + soff], src);
    };
    auto issue_m = [&](int buf, int kbase) {            // 4 ops
        const float* mp = mask + ((size_t)(b * NHEAD + h) * NQ + qblk * 128 + mrow) * NQ
                               + kbase + mseg;
        #pragma unroll
        for (int j = 0; j < 4; ++j)
            async_b128(&Mbuf[buf][mrow * 32 + mseg + j * 4], mp + j * 4);
    };

    // A operand for S-gemm: Q rows, contraction over dk (2 ops: dk 0..31, 32..63)
    V16 aq[2];
    {
        const h16* qp = Q + (size_t)(b * NQ + qbase + lm) * D_MODEL + h * DK;
        int c0 = lh ? 8 : 0;
        aq[0].q[0] = *(const u32x4*)(qp + c0);
        aq[0].q[1] = *(const u32x4*)(qp + c0 + 16);
        aq[1].q[0] = *(const u32x4*)(qp + c0 + 32);
        aq[1].q[1] = *(const u32x4*)(qp + c0 + 48);
    }

    float row_m[8], row_l[8];
    #pragma unroll
    for (int r = 0; r < 8; ++r) { row_m[r] = -1e30f; row_l[r] = 0.f; }
    v8f acc[4] = {};
    h16* pw = Pex + wave * 16 * 40;

    issue_k(0, 0);       // prologue prefetch of tile 0 (K + mask): 5 ops
    issue_m(0, 0);

    for (int kt = 0; kt < 33; ++kt) {
        const int kbase = kt * 32;
        const int cur = kt & 1;
        __syncthreads();                       // all waves done with prev buffers + Vt
        if (kt + 1 < 32)       { issue_k(cur ^ 1, kbase + 32); issue_m(cur ^ 1, kbase + 32); }
        else if (kt + 1 == 32) { issue_k(cur ^ 1, kbase + 32); }   // pad tile: no mask

        {   // synchronous V stage (transposed scatter), overlaps async stream
            size_t srow = (size_t)(b * NKE + kbase + skey) * D_MODEL + h * DK + soff;
            V16 vv; vv.q[0] = *(const u32x4*)(V + srow);
            #pragma unroll
            for (int j = 0; j < 8; ++j) Vt[(soff + j) * 40 + skey] = vv.h[j];
        }
        // drain current tile's async ops; leave next tile's in flight
        if (kt < 31)       ASYNC_WAIT(0x5);
        else if (kt == 31) ASYNC_WAIT(0x1);
        else               ASYNC_WAIT(0x0);
        __syncthreads();                       // publish Kbuf/Mbuf[cur] + Vt

        // S = Q @ K^T : two 16x16 key blocks, contraction dk=64 (2 WMMA each)
        v8f s0 = {}, s1 = {};
        {
            int kk0 = lh ? 8 : 0;
            const h16* kb = Kbuf[cur];
            #pragma unroll
            for (int hh = 0; hh < 2; ++hh) {
                V16 b0, b1;
                const u32x4* p0 = (const u32x4*)(kb + lm * 64 + hh * 32 + kk0);
                b0.q[0] = p0[0]; b0.q[1] = p0[2];
                const u32x4* p1 = (const u32x4*)(kb + (16 + lm) * 64 + hh * 32 + kk0);
                b1.q[0] = p1[0]; b1.q[1] = p1[2];
                s0 = wmma_f16(aq[hh].v, b0.v, s0);
                s1 = wmma_f16(aq[hh].v, b1.v, s1);
            }
        }

        // online softmax (row = r + 8*lh, cols striped over 16 lanes), DPP reduce
        const float* mb = Mbuf[cur];
        #pragma unroll
        for (int r = 0; r < 8; ++r) {
            int ql = wave * 16 + r + lh * 8;   // q row within the 128-row block
            float m0, m1;
            if (kt < 32) {
                m0 = mb[ql * 32 + lm];
                m1 = mb[ql * 32 + 16 + lm];
            } else {
                m0 = (1024 + lm < 1032) ? 0.f : -1e30f;  // memory slots unmasked
                m1 = -1e30f;                              // zero padding
            }
            float v0 = s0[r] * 0.125f + m0;  // 1/sqrt(DK)
            float v1 = s1[r] * 0.125f + m1;
            float tm = red_max16(fmaxf(v0, v1));
            float nm = fmaxf(row_m[r], tm);
            float sc = __expf(row_m[r] - nm);
            row_m[r] = nm;
            float p0 = __expf(v0 - nm), p1 = __expf(v1 - nm);
            float ps = red_sum16(p0 + p1);
            row_l[r] = row_l[r] * sc + ps;
            acc[0][r] *= sc; acc[1][r] *= sc; acc[2][r] *= sc; acc[3][r] *= sc;
            int prow = r + lh * 8;   // C-layout -> LDS row-major P tile
            pw[prow * 40 + lm]      = (h16)p0;
            pw[prow * 40 + 16 + lm] = (h16)p1;
        }
        // same-wave DS ops are in-order; wait so A-layout reload sees all lanes' P
        asm volatile("s_wait_dscnt 0x0" ::: "memory");

        V16 ap;
        {
            int pk0 = lh ? 8 : 0;
            ap.q[0] = *(const u32x4*)(pw + lm * 40 + pk0);
            ap.q[1] = *(const u32x4*)(pw + lm * 40 + pk0 + 16);
        }
        {
            int kk0 = lh ? 8 : 0;
            #pragma unroll
            for (int cb = 0; cb < 4; ++cb) {
                V16 bv;
                const u32x4* p = (const u32x4*)(Vt + (cb * 16 + lm) * 40 + kk0);
                bv.q[0] = p[0]; bv.q[1] = p[2];
                acc[cb] = wmma_f16(ap.v, bv.v, acc[cb]);
            }
        }
    }

    #pragma unroll
    for (int cb = 0; cb < 4; ++cb)
        #pragma unroll
        for (int r = 0; r < 8; ++r) {
            int q = qbase + r + lh * 8;
            float v = acc[cb][r] / row_l[r];
            O[(size_t)(b * NQ + q) * D_MODEL + h * DK + cb * 16 + lm] = (h16)v;
        }
}

// ------------------------------ launcher -----------------------------------

extern "C" void kernel_launch(void* const* d_in, const int* in_sizes, int n_in,
                              void* d_out, int out_size, void* d_ws, size_t ws_size,
                              hipStream_t stream) {
    const float* queries = (const float*)d_in[0];
    const float* keys    = (const float*)d_in[1];
    const float* values  = (const float*)d_in[2];
    const float* m_k     = (const float*)d_in[3];
    const float* m_v     = (const float*)d_in[4];
    const float* mask    = (const float*)d_in[5];
    const float* w_in    = (const float*)d_in[6];
    const float* b_in    = (const float*)d_in[7];
    const float* w_out   = (const float*)d_in[8];
    const float* b_out   = (const float*)d_in[9];

    char* ws = (char*)d_ws;
    const size_t OFF_W16 = 0;                         // 3M halves
    const size_t OFF_WO  = OFF_W16 + 6291456;         // 1M halves
    const size_t OFF_XQ  = OFF_WO  + 2097152;         // 8M halves (reused as O16)
    const size_t OFF_XK  = OFF_XQ  + 16777216;
    const size_t OFF_XV  = OFF_XK  + 16777216;
    const size_t OFF_Q   = OFF_XV  + 16777216;
    const size_t OFF_KX  = OFF_Q   + 16777216;        // 8*1056*1024 halves
    const size_t OFF_VX  = OFF_KX  + 17301504;

    h16* W16 = (h16*)(ws + OFF_W16);
    h16* WO  = (h16*)(ws + OFF_WO);
    h16* XQ  = (h16*)(ws + OFF_XQ);
    h16* XK  = (h16*)(ws + OFF_XK);
    h16* XV  = (h16*)(ws + OFF_XV);
    h16* Q16 = (h16*)(ws + OFF_Q);
    h16* KX  = (h16*)(ws + OFF_KX);
    h16* VX  = (h16*)(ws + OFF_VX);
    h16* O16 = XQ;   // XQ consumed by the Q projection before attention writes O16

    // 1) weight casts
    cast_f32_f16<<<(3 * 1024 * 1024) / 256, 256, 0, stream>>>(w_in, W16, 3 * 1024 * 1024);
    cast_f32_f16<<<(1024 * 1024) / 256, 256, 0, stream>>>(w_out, WO, 1024 * 1024);
    // 2) activation casts with (seq,batch)->(batch,seq) fold
    cast_act<<<(BATCH * NQ * D_MODEL) / 256, 256, 0, stream>>>(queries, XQ);
    cast_act<<<(BATCH * NQ * D_MODEL) / 256, 256, 0, stream>>>(keys, XK);
    cast_act<<<(BATCH * NQ * D_MODEL) / 256, 256, 0, stream>>>(values, XV);
    // 3) memory slot + pad rows
    fill_mem<<<(BATCH * 32 * D_MODEL) / 256, 256, 0, stream>>>(m_k, m_v, KX, VX);

    // 4) QKV projections (M=8192, N=1024, K=1024)
    dim3 ggrid(8192 / 128, D_MODEL / 128);
    gemm_wmma<0><<<ggrid, 256, 0, stream>>>(XQ, W16, b_in, (void*)Q16);
    gemm_wmma<1><<<ggrid, 256, 0, stream>>>(XK, W16 + 1048576, b_in + 1024, (void*)KX);
    gemm_wmma<1><<<ggrid, 256, 0, stream>>>(XV, W16 + 2097152, b_in + 2048, (void*)VX);

    // 5) fused flash attention over 1032 (padded 1056) keys
    attn_flash<<<dim3(NQ / 128, NHEAD, BATCH), 256, 0, stream>>>(Q16, KX, VX, mask, O16);

    // 6) output projection, fused transpose back to (seq, batch, d), fp32 out
    gemm_wmma<2><<<ggrid, 256, 0, stream>>>(O16, WO, b_out, d_out);
}
